// L2lDecoder_58952721105256
// MI455X (gfx1250) — compile-verified, matrix operand order checked by reading, not run on hardware
//
#include <hip/hip_runtime.h>
#include <hip/hip_bf16.h>

// ---------------------------------------------------------------------------
// L2L decoder for MI455X (gfx1250, wave32).
// GEMMs + attention use v_wmma_f32_16x16x32_f16 (f16 in, f32 accumulate).
// GEMM: block = 8 waves covering 32 rows x 512 cols; A panel staged in LDS
// as f16 in 32x512 chunks (32 KB); each wave computes a 32x64 tile with 8
// accumulators -> 8 WMMAs per k-step off 2 A frags + 4 shared B frags.
// Model dims: B=16, TIN=64, D=512, H=8, DH=64, FF=2048, NL=6, K=5, OUT=128,
// T = TIN*8 = 512, M = B*T = 8192.
// ---------------------------------------------------------------------------

typedef __attribute__((ext_vector_type(16))) _Float16 v16h;
typedef __attribute__((ext_vector_type(8)))  _Float16 h8;
typedef __attribute__((ext_vector_type(8)))  float    v8f;

#define Dm   512
#define Tm   512
#define Bm   16
#define Hm   8
#define DHm  64
#define FFm  2048
#define NLm  6
#define Mrows (Bm * Tm)
#define KC   512   // K-chunk staged in LDS per GEMM iteration

// ----- WMMA helpers ---------------------------------------------------------

__device__ __forceinline__ v8f wmma_f16(v16h a, v16h b, v8f c) {
  // 8 args: (neg_a, A, neg_b, B, c_mod, C, reuse_a, reuse_b)
  return __builtin_amdgcn_wmma_f32_16x16x32_f16(false, a, false, b,
                                                (short)0, c, false, false);
}

// 16-bit A/B fragment lane->K mapping (ISA 7.12.2):
// lanes 0-15 : VGPR0-3 hold K=0..7,  VGPR4-7 hold K=16..23
// lanes 16-31: VGPR0-3 hold K=8..15, VGPR4-7 hold K=24..31
__device__ __forceinline__ int kmap16(int i, int g) {
  return (i < 8 ? i : i + 8) + g * 8;
}

// Fragment from f16 memory where this lane's halves are two contiguous
// 8-half (16 B) chunks at p and p+16: forces b128 loads (global or LDS).
__device__ __forceinline__ v16h frag_h8(const _Float16* p) {
  h8 lo = *(const h8*)(p);
  h8 hi = *(const h8*)(p + 16);
  v16h f;
#pragma unroll
  for (int i = 0; i < 8; ++i) {
    f[i] = lo[i];
    f[i + 8] = hi[i];
  }
  return f;
}

// Generic fragment gather from f32: element[i] = base[r*rstride + kmap*kstride]
template <typename TS>
__device__ __forceinline__ v16h load_frag(const TS* base, long rstride,
                                          long kstride, int lane) {
  const int g = lane >> 4;
  const int r = lane & 15;
  v16h f;
#pragma unroll
  for (int i = 0; i < 16; ++i) {
    f[i] = (_Float16)base[(long)r * rstride + (long)kmap16(i, g) * kstride];
  }
  return f;
}

// ----- weight conversion ----------------------------------------------------

__global__ void f32_to_f16_kernel(const float* __restrict__ in,
                                  _Float16* __restrict__ out, int n) {
  int i = blockIdx.x * blockDim.x + threadIdx.x;
  if (i < n) out[i] = (_Float16)in[i];
}

// ----- expander -------------------------------------------------------------

__device__ __forceinline__ float leaky(float x) {
  return x >= 0.f ? x : 0.2f * x;
}

// ConvTranspose1d(k=5, stride=2, pad=2, out_pad=1): Tin=64 -> L=128,
// then bias -> leaky -> BN(stage 0).   in: (B, TIN, D)  out: (B, D, 128)
__global__ void deconv_kernel(const float* __restrict__ in,
                              const float* __restrict__ w,   // (Din, Dout, 5)
                              const float* __restrict__ bias,
                              const float* __restrict__ bng,
                              const float* __restrict__ bnb,
                              const float* __restrict__ bnm,
                              const float* __restrict__ bnv,
                              float* __restrict__ out) {
  int idx = blockIdx.x * blockDim.x + threadIdx.x;
  if (idx >= Bm * Dm * 128) return;
  int t  = idx & 127;
  int co = (idx >> 7) & (Dm - 1);
  int b  = idx >> 16;
  float acc = bias[co];
  // t = 2*ti - 2 + k  ->  ti = (t + 2 - k)/2, k parity == t parity
  for (int k = (t & 1); k < 5; k += 2) {
    int num = t + 2 - k;
    if (num < 0) continue;
    int ti = num >> 1;
    if (ti >= 64) continue;
    const float* xp = in + ((long)b * 64 + ti) * Dm;
    const float* wp = w + (long)co * 5 + k;
    for (int ci = 0; ci < Dm; ++ci) acc += xp[ci] * wp[(long)ci * Dm * 5];
  }
  float y = leaky(acc);
  y = (y - bnm[co]) * (bng[co] * rsqrtf(bnv[co] + 1e-5f)) + bnb[co];
  out[((long)b * Dm + co) * 128 + t] = y;
}

// Conv1d(k=5, replicate pad 2) -> bias -> leaky -> BN(stage) -> repeat x2
// in: (B, D, Lin)  out: (B, D, 2*Lin)
__global__ void conv_stage_kernel(const float* __restrict__ in,
                                  const float* __restrict__ w,  // (Dout, Din, 5)
                                  const float* __restrict__ bias,
                                  const float* __restrict__ bng,
                                  const float* __restrict__ bnb,
                                  const float* __restrict__ bnm,
                                  const float* __restrict__ bnv,
                                  float* __restrict__ out, int Lin) {
  int idx = blockIdx.x * blockDim.x + threadIdx.x;
  if (idx >= Bm * Dm * Lin) return;
  int t  = idx % Lin;
  int co = (idx / Lin) & (Dm - 1);
  int b  = idx / (Lin * Dm);
  float acc = bias[co];
#pragma unroll
  for (int k = 0; k < 5; ++k) {
    int pos = t + k - 2;
    pos = pos < 0 ? 0 : (pos >= Lin ? Lin - 1 : pos);
    const float* xp = in + (long)b * Dm * Lin + pos;
    const float* wp = w + (long)co * Dm * 5 + k;
    for (int ci = 0; ci < Dm; ++ci) acc += xp[(long)ci * Lin] * wp[(long)ci * 5];
  }
  float y = leaky(acc);
  y = (y - bnm[co]) * (bng[co] * rsqrtf(bnv[co] + 1e-5f)) + bnb[co];
  float* op = out + ((long)b * Dm + co) * (2 * Lin) + 2 * t;
  op[0] = y;
  op[1] = y;
}

// (B, D, T) channel-major -> (B*T, D) token-major
__global__ void transpose_kernel(const float* __restrict__ in,
                                 float* __restrict__ out) {
  int idx = blockIdx.x * blockDim.x + threadIdx.x;  // ((b*T + t)*D + ci)
  if (idx >= Bm * Tm * Dm) return;
  int ci = idx & (Dm - 1);
  int t  = (idx >> 9) & (Tm - 1);
  int b  = idx >> 18;
  out[idx] = in[((long)b * Dm + ci) * Tm + t];
}

// ----- WMMA GEMM:  C[M,N] = A[M,K] * W[N,K]^T + bias  (+ optional exact GELU)
// Block = 8 waves, 32 rows x 512 cols. A staged in LDS as f16, 32x512 chunk
// (32 KB) at a time. Each wave: 32x64 tile = 8 accumulators; per k-step:
// 2 A frags (LDS) + 4 B frags (global f16) feed 8 WMMAs.
__global__ __launch_bounds__(256) void gemm_wmma_kernel(
    const float* __restrict__ A, const _Float16* __restrict__ W,
    const float* __restrict__ bias, float* __restrict__ C, int M, int N,
    int Kd, int act) {
  __shared__ _Float16 sA[32 * KC];  // 32 KB
  const int tid = threadIdx.x;
  const int lane = tid & 31;
  const int wave = tid >> 5;
  const int row0 = blockIdx.y * 32;
  const int col0 = blockIdx.x * 512 + wave * 64;

  const int g = lane >> 4;
  const int r = lane & 15;  // row (m) within 16-tile / B row (n) within tile
  const _Float16* Wb = W + (long)col0 * Kd;

  v8f acc0[4] = {};
  v8f acc1[4] = {};

  for (int kc = 0; kc < Kd; kc += KC) {
    // cooperative A-panel stage: 32 rows x KC cols fp32 -> f16 LDS
    __syncthreads();
    for (int i = tid * 4; i < 32 * KC; i += 256 * 4) {
      int rr = i >> 9;          // i / KC
      int cc = i & (KC - 1);    // i % KC
      float4 v4 = *(const float4*)(A + (long)(row0 + rr) * Kd + kc + cc);
      sA[i + 0] = (_Float16)v4.x;
      sA[i + 1] = (_Float16)v4.y;
      sA[i + 2] = (_Float16)v4.z;
      sA[i + 3] = (_Float16)v4.w;
    }
    __syncthreads();

    for (int k0 = 0; k0 < KC; k0 += 32) {
      v16h a0 = frag_h8(sA + (long)r * KC + k0 + g * 8);
      v16h a1 = frag_h8(sA + (long)(16 + r) * KC + k0 + g * 8);
      // stream-prefetch next K-step of this lane's weight row
      __builtin_prefetch((const void*)(Wb + (long)r * Kd + kc + k0 + 32), 0, 1);
#pragma unroll
      for (int t = 0; t < 4; ++t) {
        v16h b = frag_h8(Wb + (long)(t * 16 + r) * Kd + kc + k0 + g * 8);
        acc0[t] = wmma_f16(a0, b, acc0[t]);
        acc1[t] = wmma_f16(a1, b, acc1[t]);
      }
    }
  }

  const float* __restrict__ bp = bias;
#pragma unroll
  for (int t = 0; t < 4; ++t) {
    int col = col0 + t * 16 + r;
    float bv = bp ? bp[col] : 0.f;
#pragma unroll
    for (int rr = 0; rr < 8; ++rr) {
      int row = row0 + rr + 8 * g;  // f32 C/D layout (ISA 7.12.2)
      float v0 = acc0[t][rr] + bv;
      float v1 = acc1[t][rr] + bv;
      if (act == 1) {
        v0 = 0.5f * v0 * (1.f + erff(v0 * 0.70710678118654752f));
        v1 = 0.5f * v1 * (1.f + erff(v1 * 0.70710678118654752f));
      }
      C[(long)row * N + col] = v0;
      C[(long)(row + 16) * N + col] = v1;
    }
  }
}

// ----- attention: one wave per (b, h, 16-row Q tile) ------------------------
// qkv: (B*T, 3*D) rows; q at col h*64, k at 512+h*64, v at 1024+h*64.
__global__ __launch_bounds__(32) void attn_kernel(
    const float* __restrict__ qkv, float* __restrict__ o) {
  __shared__ float sS[16 * Tm];  // 16 x 512 fp32 score tile = 32 KB LDS
  const int lane = threadIdx.x;
  const int qt = blockIdx.x, h = blockIdx.y, b = blockIdx.z;
  const int row0 = qt * 16;

  const float* base = qkv + (long)b * Tm * (3 * Dm);
  const float* Qp = base + h * DHm;
  const float* Kp = base + Dm + h * DHm;
  const float* Vp = base + 2 * Dm + h * DHm;

  const float slope = exp2f(-(float)(h + 1));
  const float scale = 0.125f;  // 1/sqrt(64)
  const int g = lane >> 4;
  const int n = lane & 15;

  // Q fragments for the two DH=64 halves (K chunks of 32)
  v16h aq0 = load_frag<float>(Qp + (long)row0 * (3 * Dm),      3 * Dm, 1, lane);
  v16h aq1 = load_frag<float>(Qp + (long)row0 * (3 * Dm) + 32, 3 * Dm, 1, lane);

  // S = Q K^T * scale + alibi  (16 x 512), WMMA contraction over DH
  for (int j = 0; j < Tm / 16; ++j) {
    v16h bk0 = load_frag<float>(Kp + (long)(j * 16) * (3 * Dm),      3 * Dm, 1, lane);
    v16h bk1 = load_frag<float>(Kp + (long)(j * 16) * (3 * Dm) + 32, 3 * Dm, 1, lane);
    v8f acc = {};
    acc = wmma_f16(aq0, bk0, acc);
    acc = wmma_f16(aq1, bk1, acc);
#pragma unroll
    for (int r = 0; r < 8; ++r) {
      int m = r + 8 * g;
      int col = j * 16 + n;
      float ab = -slope * fabsf((float)(row0 + m - col));
      sS[m * Tm + col] = acc[r] * scale + ab;
    }
  }
  __syncthreads();

  // row softmax (lanes 0..15 own one row each)
  if (lane < 16) {
    float* row = sS + lane * Tm;
    float mx = -3.0e38f;
    for (int j = 0; j < Tm; ++j) mx = fmaxf(mx, row[j]);
    float sum = 0.f;
    for (int j = 0; j < Tm; ++j) {
      float e = expf(row[j] - mx);
      row[j] = e;
      sum += e;
    }
    float inv = 1.f / sum;
    for (int j = 0; j < Tm; ++j) row[j] *= inv;
  }
  __syncthreads();

  // O = P V  (16 x 64): one P fragment feeds all 4 DH tiles
  v8f oacc[4] = {};
  for (int j = 0; j < Tm / 32; ++j) {
    v16h ap = load_frag<float>(sS + j * 32, Tm, 1, lane);  // P rows from LDS
#pragma unroll
    for (int dt = 0; dt < 4; ++dt) {
      v16h bv = load_frag<float>(Vp + (long)(j * 32) * (3 * Dm) + dt * 16,
                                 1, 3 * Dm, lane);          // V columns
      oacc[dt] = wmma_f16(ap, bv, oacc[dt]);
    }
  }
#pragma unroll
  for (int dt = 0; dt < 4; ++dt) {
#pragma unroll
    for (int r = 0; r < 8; ++r) {
      int m = r + 8 * g;
      o[((long)b * Tm + row0 + m) * Dm + h * DHm + dt * 16 + n] = oacc[dt][r];
    }
  }
}

// ----- fused residual add + LayerNorm (row = one token, D=512) --------------
__global__ __launch_bounds__(128) void add_ln_kernel(
    const float* __restrict__ x, const float* __restrict__ o,
    const float* __restrict__ gam, const float* __restrict__ bet,
    float* __restrict__ y) {
  const int row = blockIdx.x;
  const int tid = threadIdx.x;
  __shared__ float red[128];
  const float* xr = x + (long)row * Dm;
  const float* orr = o + (long)row * Dm;
  float v[4];
  float s = 0.f;
#pragma unroll
  for (int i = 0; i < 4; ++i) {
    int c = tid + i * 128;
    v[i] = xr[c] + orr[c];
    s += v[i];
  }
  red[tid] = s;
  __syncthreads();
  for (int off = 64; off >= 1; off >>= 1) {
    if (tid < off) red[tid] += red[tid + off];
    __syncthreads();
  }
  float mean = red[0] * (1.f / Dm);
  __syncthreads();
  float vs = 0.f;
#pragma unroll
  for (int i = 0; i < 4; ++i) {
    float d = v[i] - mean;
    vs += d * d;
  }
  red[tid] = vs;
  __syncthreads();
  for (int off = 64; off >= 1; off >>= 1) {
    if (tid < off) red[tid] += red[tid + off];
    __syncthreads();
  }
  float inv = rsqrtf(red[0] * (1.f / Dm) + 1e-5f);
  __syncthreads();
#pragma unroll
  for (int i = 0; i < 4; ++i) {
    int c = tid + i * 128;
    y[(long)row * Dm + c] = (v[i] - mean) * inv * gam[c] + bet[c];
  }
}

// ----- final smooth conv: (M, D) tokens -> (B, T, OUT=128), zero pad k=5 ----
__global__ void smooth_kernel(const float* __restrict__ x,
                              const float* __restrict__ w,   // (128, 512, 5)
                              const float* __restrict__ bias,
                              float* __restrict__ out) {
  int idx = blockIdx.x * blockDim.x + threadIdx.x;  // m*128 + o
  if (idx >= Mrows * 128) return;
  int oc = idx & 127;
  int m  = idx >> 7;
  int t  = m & (Tm - 1);
  int b  = m >> 9;
  float acc = bias[oc];
#pragma unroll
  for (int k = 0; k < 5; ++k) {
    int tt = t + k - 2;
    if (tt < 0 || tt >= Tm) continue;
    const float* xp = x + ((long)b * Tm + tt) * Dm;
    const float* wp = w + (long)oc * Dm * 5 + k;
    for (int ci = 0; ci < Dm; ++ci) acc += xp[ci] * wp[(long)ci * 5];
  }
  out[idx] = acc;
}

// ---------------------------------------------------------------------------

extern "C" void kernel_launch(void* const* d_in, const int* in_sizes, int n_in,
                              void* d_out, int out_size, void* d_ws,
                              size_t ws_size, hipStream_t stream) {
  const float* inputs   = (const float*)d_in[0];
  const float* deconv_w = (const float*)d_in[1];
  const float* deconv_b = (const float*)d_in[2];
  const float* bn_gamma = (const float*)d_in[3];
  const float* bn_beta  = (const float*)d_in[4];
  const float* bn_mean  = (const float*)d_in[5];
  const float* bn_var   = (const float*)d_in[6];
  const float* conv_w   = (const float*)d_in[7];
  const float* conv_b   = (const float*)d_in[8];
  const float* emb_w    = (const float*)d_in[9];
  const float* emb_b    = (const float*)d_in[10];
  const float* qkv_w    = (const float*)d_in[11];
  const float* qkv_b    = (const float*)d_in[12];
  const float* outp_w   = (const float*)d_in[13];
  const float* outp_b   = (const float*)d_in[14];
  const float* ln1_g    = (const float*)d_in[15];
  const float* ln1_b    = (const float*)d_in[16];
  const float* ff1_w    = (const float*)d_in[17];
  const float* ff1_b    = (const float*)d_in[18];
  const float* ff2_w    = (const float*)d_in[19];
  const float* ff2_b    = (const float*)d_in[20];
  const float* ln2_g    = (const float*)d_in[21];
  const float* ln2_b    = (const float*)d_in[22];
  const float* smooth_w = (const float*)d_in[23];
  const float* smooth_b = (const float*)d_in[24];

  // ---- workspace carve-out -------------------------------------------------
  char* ws = (char*)d_ws;
  size_t off = 0;
  auto carve = [&](size_t bytes) -> void* {
    void* p = (void*)(ws + off);
    off += (bytes + 255) & ~(size_t)255;
    return p;
  };
  _Float16* embWh = (_Float16*)carve(sizeof(_Float16) * Dm * Dm);
  _Float16* qkvWh = (_Float16*)carve(sizeof(_Float16) * NLm * 3 * Dm * Dm);
  _Float16* outWh = (_Float16*)carve(sizeof(_Float16) * NLm * Dm * Dm);
  _Float16* ff1Wh = (_Float16*)carve(sizeof(_Float16) * NLm * FFm * Dm);
  _Float16* ff2Wh = (_Float16*)carve(sizeof(_Float16) * NLm * Dm * FFm);
  float* e0    = (float*)carve(sizeof(float) * Bm * Dm * 128);
  float* e1    = (float*)carve(sizeof(float) * Bm * Dm * 256);
  float* e2    = (float*)carve(sizeof(float) * Bm * Dm * Tm);
  float* xbuf  = (float*)carve(sizeof(float) * Mrows * Dm);
  float* tmp   = (float*)carve(sizeof(float) * Mrows * Dm);
  float* attno = (float*)carve(sizeof(float) * Mrows * Dm);
  float* qkvb  = (float*)carve(sizeof(float) * Mrows * 3 * Dm);
  float* ffb   = (float*)carve(sizeof(float) * Mrows * FFm);

  auto cvt = [&](const float* src, _Float16* dst, int n) {
    f32_to_f16_kernel<<<(n + 255) / 256, 256, 0, stream>>>(src, dst, n);
  };
  cvt(emb_w,  embWh, Dm * Dm);
  cvt(qkv_w,  qkvWh, NLm * 3 * Dm * Dm);
  cvt(outp_w, outWh, NLm * Dm * Dm);
  cvt(ff1_w,  ff1Wh, NLm * FFm * Dm);
  cvt(ff2_w,  ff2Wh, NLm * Dm * FFm);

  // ---- expander ------------------------------------------------------------
  {
    int n = Bm * Dm * 128;
    deconv_kernel<<<(n + 255) / 256, 256, 0, stream>>>(
        inputs, deconv_w, deconv_b, bn_gamma, bn_beta, bn_mean, bn_var, e0);
  }
  {
    int n = Bm * Dm * 128;
    conv_stage_kernel<<<(n + 255) / 256, 256, 0, stream>>>(
        e0, conv_w, conv_b, bn_gamma + Dm, bn_beta + Dm, bn_mean + Dm,
        bn_var + Dm, e1, 128);
  }
  {
    int n = Bm * Dm * 256;
    conv_stage_kernel<<<(n + 255) / 256, 256, 0, stream>>>(
        e1, conv_w + (long)Dm * Dm * 5, conv_b + Dm, bn_gamma + 2 * Dm,
        bn_beta + 2 * Dm, bn_mean + 2 * Dm, bn_var + 2 * Dm, e2, 256);
  }
  {
    int n = Bm * Tm * Dm;
    transpose_kernel<<<(n + 255) / 256, 256, 0, stream>>>(e2, tmp);
  }

  auto gemm = [&](const float* A, const _Float16* W, const float* bias,
                  float* C, int M, int N, int Kd, int act) {
    dim3 grid(N / 512, M / 32);
    gemm_wmma_kernel<<<grid, 256, 0, stream>>>(A, W, bias, C, M, N, Kd, act);
  };

  // ---- embedding -----------------------------------------------------------
  gemm(tmp, embWh, emb_b, xbuf, Mrows, Dm, Dm, 0);

  // ---- transformer layers --------------------------------------------------
  for (int l = 0; l < NLm; ++l) {
    gemm(xbuf, qkvWh + (size_t)l * 3 * Dm * Dm, qkv_b + l * 3 * Dm, qkvb,
         Mrows, 3 * Dm, Dm, 0);
    dim3 ag(Tm / 16, Hm, Bm);
    attn_kernel<<<ag, 32, 0, stream>>>(qkvb, attno);
    gemm(attno, outWh + (size_t)l * Dm * Dm, outp_b + l * Dm, tmp, Mrows, Dm,
         Dm, 0);
    add_ln_kernel<<<Mrows, 128, 0, stream>>>(xbuf, tmp, ln1_g + l * Dm,
                                             ln1_b + l * Dm, xbuf);
    gemm(xbuf, ff1Wh + (size_t)l * FFm * Dm, ff1_b + l * FFm, ffb, Mrows, FFm,
         Dm, 1);
    gemm(ffb, ff2Wh + (size_t)l * Dm * FFm, ff2_b + l * Dm, tmp, Mrows, Dm,
         FFm, 0);
    add_ln_kernel<<<Mrows, 128, 0, stream>>>(xbuf, tmp, ln2_g + l * Dm,
                                             ln2_b + l * Dm, xbuf);
  }

  // ---- smooth conv ---------------------------------------------------------
  {
    int n = Mrows * 128;
    smooth_kernel<<<(n + 255) / 256, 256, 0, stream>>>(xbuf, smooth_w,
                                                       smooth_b,
                                                       (float*)d_out);
  }
}
